// RecurrentCorrector_44306882626012
// MI455X (gfx1250) — matrix operand order repaired and checked by reading, not compiled, not run on hardware
//
#include <hip/hip_runtime.h>
#include <cstdint>

// ---------------------------------------------------------------------------
// MI455X / gfx1250 GRU decoder, bf16-resident activations, pointer-marched
// WMMA inner loops.
//   Phase A: GI0 = y @ Wih0^T + bih0            (WMMA GEMM, M=8192,K=512,N=3072)
//   Phase B: 256 seq. steps: layer0 (hidden GEMM + gates),
//            layer1 (input+hidden GEMM + gates) -> rnn_out (bf16)
//   Phase C: GIc = [y_states, rnn_out] @ Wih_c^T + out_bih (dual-A GEMM, K=2048)
//   Phase D: 256 seq. steps: out-GRU (prev-feedback + hidden GEMM + gates),
//            proj GEMM + tanh -> d_out (fp32) + bf16 prev buffer.
// ---------------------------------------------------------------------------

typedef __attribute__((ext_vector_type(16))) __bf16 v16bf;
typedef __attribute__((ext_vector_type(8)))  float  v8f;

#define WMMA_BF16(a, b, c) \
  __builtin_amdgcn_wmma_f32_16x16x32_bf16(false, (a), false, (b), (short)0, (c), false, false)

static __device__ __forceinline__ unsigned short f2bf(float f) {
  unsigned u = __float_as_uint(f);
  u += 0x7fffu + ((u >> 16) & 1u);      // round-to-nearest-even
  return (unsigned short)(u >> 16);
}

static __device__ __forceinline__ float sigmoidf_(float x) {
  return 1.0f / (1.0f + __expf(-x));
}

// Load a 16x32 bf16 A fragment from a per-lane pointer (two b128 loads).
static __device__ __forceinline__ v16bf ld_a(const unsigned short* __restrict__ p) {
  union { v16bf v; uint4 q[2]; } f;
  f.q[0] = *(const uint4*)p;
  f.q[1] = *(const uint4*)(p + 16);
  return f.v;
}
// Load a packed B fragment from a per-lane pointer (32 contiguous bytes).
static __device__ __forceinline__ v16bf ld_b(const unsigned short* __restrict__ p) {
  return *(const v16bf*)p;
}

// Per-lane A base pointer for M-tile at mBase, k=0.
static __device__ __forceinline__ const unsigned short*
a_base(const unsigned short* A, long lda, int mBase) {
  int lane = threadIdx.x & 31;
  return A + (long)(mBase + (lane & 15)) * lda + (lane >> 4) * 8;
}
// Per-lane packed-B base pointer for N-tile nt, k-chunk 0. Frag stride = 512 halfs.
static __device__ __forceinline__ const unsigned short*
b_base(const unsigned short* P, int nt, int kChunks) {
  int lane = threadIdx.x & 31;
  return P + (long)nt * kChunks * 512 + (long)lane * 16;
}

// ---------------------------------------------------------------------------
// Elementwise fp32 -> bf16 (one-shot for y, y_states).
// ---------------------------------------------------------------------------
__global__ void cvt_f32_bf16(const float* __restrict__ src,
                             unsigned short* __restrict__ dst, long n) {
  long i = ((long)blockIdx.x * blockDim.x + threadIdx.x) * 4;
  if (i + 3 < n) {
#pragma unroll
    for (int k = 0; k < 4; ++k) dst[i + k] = f2bf(src[i + k]);
  } else {
    for (long k = i; k < n; ++k) dst[k] = f2bf(src[k]);
  }
}

// ---------------------------------------------------------------------------
// Pack W (fp32 row-major, rows = output cols, row stride ldw, K segment at
// kOff of length Kseg) into bf16 B-fragment order. One-shot per launch.
// Lane l of fragment (nt,kt): column n = nt*16 + (l&15),
// k = kt*32 + (l>>4)*8 + [0..7], then kt*32 + 16 + (l>>4)*8 + [0..7].
// ---------------------------------------------------------------------------
__global__ void pack_w_bf16(const float* __restrict__ W, long ldw, long kOff,
                            int Kseg, int Nrows, unsigned short* __restrict__ outp) {
  int tid = blockIdx.x * blockDim.x + threadIdx.x;
  int kChunks = Kseg >> 5;
  int total = (Nrows >> 4) * kChunks * 32;
  if (tid >= total) return;
  int lane = tid & 31;
  int frag = tid >> 5;
  int kt = frag % kChunks;
  int nt = frag / kChunks;
  int n  = nt * 16 + (lane & 15);
  int kb = kt * 32 + (lane >> 4) * 8;
  const float* src = W + (long)n * ldw + kOff + kb;
  unsigned short* dst = outp + (size_t)tid * 16;
#pragma unroll
  for (int i = 0; i < 16; ++i) dst[i] = f2bf(src[i < 8 ? i : i + 8]);
}

// ---------------------------------------------------------------------------
// Big batched GEMM + bias: C[M,N] = [A0 | A1][M,K] @ Bpack + bias, A in bf16.
// Two sequential K loops (one per A source), all streams pointer-marched.
// Grid (N/64, M/32), block 128 (4 waves, one 16-col tile each).
// ---------------------------------------------------------------------------
__global__ void gemm_bias_wmma(const unsigned short* __restrict__ A0, long lda0,
                               const unsigned short* __restrict__ A1, long lda1,
                               int Ksplit, int K,
                               const unsigned short* __restrict__ Bp,
                               const float* __restrict__ bias,
                               float* __restrict__ C, int N) {
  int wave = threadIdx.x >> 5;
  int nt = blockIdx.x * 4 + wave;
  int mBase = blockIdx.y * 32;
  int kChunks = K >> 5;
  int k0 = Ksplit >> 5;            // chunks sourced from A0
  int k1 = kChunks - k0;           // chunks sourced from A1

  v8f acc0 = {}, acc1 = {};
  const unsigned short* bp = b_base(Bp, nt, kChunks);

  {
    const unsigned short* aLo = a_base(A0, lda0, mBase);
    const unsigned short* aHi = a_base(A0, lda0, mBase + 16);
    for (int kt = 0; kt < k0; ++kt) {
      __builtin_prefetch(bp + 512, 0, 1);
      v16bf a0 = ld_a(aLo), a1 = ld_a(aHi), b = ld_b(bp);
      acc0 = WMMA_BF16(a0, b, acc0);
      acc1 = WMMA_BF16(a1, b, acc1);
      aLo += 32; aHi += 32; bp += 512;
    }
  }
  if (k1 > 0) {
    const unsigned short* aLo = a_base(A1, lda1, mBase);
    const unsigned short* aHi = a_base(A1, lda1, mBase + 16);
    for (int kt = 0; kt < k1; ++kt) {
      __builtin_prefetch(bp + 512, 0, 1);
      v16bf a0 = ld_a(aLo), a1 = ld_a(aHi), b = ld_b(bp);
      acc0 = WMMA_BF16(a0, b, acc0);
      acc1 = WMMA_BF16(a1, b, acc1);
      aLo += 32; aHi += 32; bp += 512;
    }
  }

  int lane = threadIdx.x & 31;
  int lc = lane & 15;
  int mh = (lane >> 4) * 8;
  int col = nt * 16 + lc;
  float bv = bias ? bias[col] : 0.0f;
#pragma unroll
  for (int i = 0; i < 8; ++i) {
    C[(size_t)(mBase + mh + i) * N + col]      = acc0[i] + bv;
    C[(size_t)(mBase + 16 + mh + i) * N + col] = acc1[i] + bv;
  }
}

// ---------------------------------------------------------------------------
// One GRU timestep (B=32, H=1024), fused GEMM(s) + gates.
//   gi = (GI ? GI[t] : 0) + (HAS_X ? Xbf @ WihP + bih : 0)
//   gh = HprevBf @ WhhP + bhh
//   r = sig(gi_r+gh_r); z = sig(gi_z+gh_z); n = tanh(gi_n + r*gh_n)
//   h' = (1-z)*n + z*h  -> Hout (fp32) + HoutBf (bf16) [+ RoutBf scatter]
// HAS_X is a compile-time specialization: no control flow in the K loop.
// Grid 16 x 128 threads (64 waves, 16 cols each). K = H = 1024.
// ---------------------------------------------------------------------------
template <bool HAS_X>
__global__ void gru_step_wmma(int t, int T,
                              const float* __restrict__ GI,
                              const unsigned short* __restrict__ Xbf, long xstride,
                              const unsigned short* __restrict__ WihP,
                              const float* __restrict__ bih,
                              const float* __restrict__ Hprev,
                              const unsigned short* __restrict__ HprevBf,
                              const unsigned short* __restrict__ WhhP,
                              const float* __restrict__ bhh,
                              float* __restrict__ Hout,
                              unsigned short* __restrict__ HoutBf,
                              unsigned short* __restrict__ RoutBf) {
  const int H = 1024;
  const int kChunks = H >> 5;  // 32
  int wave = threadIdx.x >> 5;
  int j = (blockIdx.x * 4 + wave) * 16;
  int ntR = j >> 4, ntZ = (H + j) >> 4, ntN = (2 * H + j) >> 4;

  v8f hR0 = {}, hR1 = {}, hZ0 = {}, hZ1 = {}, hN0 = {}, hN1 = {};
  v8f iR0 = {}, iR1 = {}, iZ0 = {}, iZ1 = {}, iN0 = {}, iN1 = {};

  const unsigned short* hLo = a_base(HprevBf, H, 0);
  const unsigned short* hHi = a_base(HprevBf, H, 16);
  const unsigned short* brP = b_base(WhhP, ntR, kChunks);
  const unsigned short* bzP = b_base(WhhP, ntZ, kChunks);
  const unsigned short* bnP = b_base(WhhP, ntN, kChunks);
  const unsigned short* xLo = nullptr;
  const unsigned short* xHi = nullptr;
  const unsigned short* crP = nullptr;
  const unsigned short* czP = nullptr;
  const unsigned short* cnP = nullptr;
  if (HAS_X) {
    xLo = a_base(Xbf, xstride, 0);
    xHi = a_base(Xbf, xstride, 16);
    crP = b_base(WihP, ntR, kChunks);
    czP = b_base(WihP, ntZ, kChunks);
    cnP = b_base(WihP, ntN, kChunks);
  }

  for (int kt = 0; kt < kChunks; ++kt) {
    __builtin_prefetch(brP + 512, 0, 1);
    __builtin_prefetch(bnP + 512, 0, 1);
    v16bf ah0 = ld_a(hLo), ah1 = ld_a(hHi);
    v16bf br = ld_b(brP), bz = ld_b(bzP), bn = ld_b(bnP);
    hR0 = WMMA_BF16(ah0, br, hR0); hR1 = WMMA_BF16(ah1, br, hR1);
    hZ0 = WMMA_BF16(ah0, bz, hZ0); hZ1 = WMMA_BF16(ah1, bz, hZ1);
    hN0 = WMMA_BF16(ah0, bn, hN0); hN1 = WMMA_BF16(ah1, bn, hN1);
    hLo += 32; hHi += 32; brP += 512; bzP += 512; bnP += 512;
    if (HAS_X) {
      v16bf ax0 = ld_a(xLo), ax1 = ld_a(xHi);
      v16bf cr = ld_b(crP), cz = ld_b(czP), cn = ld_b(cnP);
      iR0 = WMMA_BF16(ax0, cr, iR0); iR1 = WMMA_BF16(ax1, cr, iR1);
      iZ0 = WMMA_BF16(ax0, cz, iZ0); iZ1 = WMMA_BF16(ax1, cz, iZ1);
      iN0 = WMMA_BF16(ax0, cn, iN0); iN1 = WMMA_BF16(ax1, cn, iN1);
      xLo += 32; xHi += 32; crP += 512; czP += 512; cnP += 512;
    }
  }

  int lane = threadIdx.x & 31;
  int lc = lane & 15;
  int mh = (lane >> 4) * 8;
  int col = j + lc;
  float bR = bhh[col], bZ = bhh[H + col], bN = bhh[2 * H + col];
  float biR = 0.f, biZ = 0.f, biN = 0.f;
  if (HAS_X && bih) { biR = bih[col]; biZ = bih[H + col]; biN = bih[2 * H + col]; }

  auto doTile = [&](const v8f& aiR, const v8f& aiZ, const v8f& aiN,
                    const v8f& ahR, const v8f& ahZ, const v8f& ahN, int mb) {
#pragma unroll
    for (int i = 0; i < 8; ++i) {
      int m = mb + mh + i;
      size_t gbase = ((size_t)m * T + t) * (size_t)(3 * H);
      float giR = 0.f, giZ = 0.f, giN = 0.f;
      if (GI) { giR = GI[gbase + col]; giZ = GI[gbase + H + col]; giN = GI[gbase + 2 * H + col]; }
      if (HAS_X) { giR += aiR[i] + biR; giZ += aiZ[i] + biZ; giN += aiN[i] + biN; }
      float ghR = ahR[i] + bR, ghZ = ahZ[i] + bZ, ghN = ahN[i] + bN;
      float r = sigmoidf_(giR + ghR);
      float z = sigmoidf_(giZ + ghZ);
      float n = tanhf(giN + r * ghN);
      float hp = Hprev[(size_t)m * H + col];
      float hn = (1.0f - z) * n + z * hp;
      Hout[(size_t)m * H + col] = hn;
      HoutBf[(size_t)m * H + col] = f2bf(hn);
      if (RoutBf) RoutBf[((size_t)m * T + t) * (size_t)H + col] = f2bf(hn);
    }
  };
  doTile(iR0, iZ0, iN0, hR0, hZ0, hN0, 0);
  doTile(iR1, iZ1, iN1, hR1, hZ1, hN1, 16);
}

// ---------------------------------------------------------------------------
// Projection + tanh: Out[(m*T+t)*D + col] = tanh(HcBf @ projW^T + pb).
// Writes fp32 to d_out and bf16 to PredBf (next step's prev-feedback A).
// ---------------------------------------------------------------------------
__global__ void proj_tanh_wmma(int t, int T,
                               const unsigned short* __restrict__ HcBf,
                               const unsigned short* __restrict__ Wp,
                               const float* __restrict__ pb,
                               float* __restrict__ Out,
                               unsigned short* __restrict__ PredBf) {
  const int H = 1024, D = 1024;
  const int kChunks = H >> 5;
  int wave = threadIdx.x >> 5;
  int nt = blockIdx.x * 4 + wave;
  v8f acc0 = {}, acc1 = {};
  const unsigned short* aLo = a_base(HcBf, H, 0);
  const unsigned short* aHi = a_base(HcBf, H, 16);
  const unsigned short* bp  = b_base(Wp, nt, kChunks);
  for (int kt = 0; kt < kChunks; ++kt) {
    __builtin_prefetch(bp + 512, 0, 1);
    v16bf a0 = ld_a(aLo), a1 = ld_a(aHi), b = ld_b(bp);
    acc0 = WMMA_BF16(a0, b, acc0);
    acc1 = WMMA_BF16(a1, b, acc1);
    aLo += 32; aHi += 32; bp += 512;
  }
  int lane = threadIdx.x & 31;
  int lc = lane & 15;
  int mh = (lane >> 4) * 8;
  int col = nt * 16 + lc;
  float bv = pb[col];
#pragma unroll
  for (int i = 0; i < 8; ++i) {
    int m0 = mh + i, m1 = 16 + mh + i;
    float p0 = tanhf(acc0[i] + bv);
    float p1 = tanhf(acc1[i] + bv);
    Out[((size_t)m0 * T + t) * (size_t)D + col] = p0;
    Out[((size_t)m1 * T + t) * (size_t)D + col] = p1;
    PredBf[(size_t)m0 * D + col] = f2bf(p0);
    PredBf[(size_t)m1 * D + col] = f2bf(p1);
  }
}

// ---------------------------------------------------------------------------
extern "C" void kernel_launch(void* const* d_in, const int* in_sizes, int n_in,
                              void* d_out, int out_size, void* d_ws, size_t ws_size,
                              hipStream_t stream) {
  (void)in_sizes; (void)n_in; (void)out_size; (void)ws_size;
  const int B = 32, T = 256, E = 512, H = 1024, D = 1024;
  const int TH = 3 * H;          // 3072
  const int M = B * T;           // 8192

  const float* y        = (const float*)d_in[0];
  const float* y_states = (const float*)d_in[3];
  const float* Wih0 = (const float*)d_in[4];
  const float* Whh0 = (const float*)d_in[5];
  const float* bih0 = (const float*)d_in[6];
  const float* bhh0 = (const float*)d_in[7];
  const float* Wih1 = (const float*)d_in[8];
  const float* Whh1 = (const float*)d_in[9];
  const float* bih1 = (const float*)d_in[10];
  const float* bhh1 = (const float*)d_in[11];
  const float* oWih = (const float*)d_in[12];   // [3H, 2D+H=3072]
  const float* oWhh = (const float*)d_in[13];
  const float* obih = (const float*)d_in[14];
  const float* obhh = (const float*)d_in[15];
  const float* pW   = (const float*)d_in[16];   // [D, H]
  const float* pb   = (const float*)d_in[17];
  float* out = (float*)d_out;                   // [B, T, D] fp32

  // ---- workspace carve-out (256B aligned slices) ----
  char* base = (char*)d_ws;
  size_t off = 0;
  auto wsAlloc = [&](size_t bytes) -> void* {
    void* p = base + off;
    off += bytes;
    off = (off + 255) & ~(size_t)255;
    return p;
  };
  float* GI0 = (float*)wsAlloc((size_t)M * TH * 4);            // 96 MB
  float* GIc = (float*)wsAlloc((size_t)M * TH * 4);            // 96 MB
  unsigned short* yBf      = (unsigned short*)wsAlloc((size_t)M * E * 2);
  unsigned short* ysBf     = (unsigned short*)wsAlloc((size_t)M * D * 2);
  unsigned short* rnnOutBf = (unsigned short*)wsAlloc((size_t)M * H * 2);
  unsigned short* Wih0p = (unsigned short*)wsAlloc((size_t)TH * E * 2);
  unsigned short* Whh0p = (unsigned short*)wsAlloc((size_t)TH * H * 2);
  unsigned short* Wih1p = (unsigned short*)wsAlloc((size_t)TH * H * 2);
  unsigned short* Whh1p = (unsigned short*)wsAlloc((size_t)TH * H * 2);
  unsigned short* WcP   = (unsigned short*)wsAlloc((size_t)TH * 2048 * 2);
  unsigned short* WpP   = (unsigned short*)wsAlloc((size_t)TH * H * 2);
  unsigned short* WhhOp = (unsigned short*)wsAlloc((size_t)TH * H * 2);
  unsigned short* projP = (unsigned short*)wsAlloc((size_t)D * H * 2);
  float* h0f[2] = { (float*)wsAlloc(B * H * 4), (float*)wsAlloc(B * H * 4) };
  float* h1f[2] = { (float*)wsAlloc(B * H * 4), (float*)wsAlloc(B * H * 4) };
  float* hof[2] = { (float*)wsAlloc(B * H * 4), (float*)wsAlloc(B * H * 4) };
  unsigned short* h0b[2] = { (unsigned short*)wsAlloc(B * H * 2), (unsigned short*)wsAlloc(B * H * 2) };
  unsigned short* h1b[2] = { (unsigned short*)wsAlloc(B * H * 2), (unsigned short*)wsAlloc(B * H * 2) };
  unsigned short* hob[2] = { (unsigned short*)wsAlloc(B * H * 2), (unsigned short*)wsAlloc(B * H * 2) };
  unsigned short* predBf = (unsigned short*)wsAlloc((size_t)B * D * 2);
  unsigned short* zbufBf = (unsigned short*)wsAlloc(H * 2);

  // ---- zero t=0 state (bf16 zero == 0x0000; graph-capture-safe) ----
  hipMemsetAsync(h0f[0], 0, B * H * 4, stream);
  hipMemsetAsync(h1f[0], 0, B * H * 4, stream);
  hipMemsetAsync(hof[0], 0, B * H * 4, stream);
  hipMemsetAsync(h0b[0], 0, B * H * 2, stream);
  hipMemsetAsync(h1b[0], 0, B * H * 2, stream);
  hipMemsetAsync(hob[0], 0, B * H * 2, stream);
  hipMemsetAsync(zbufBf, 0, H * 2,     stream);

  // ---- one-shot conversions and weight packing ----
  {
    long ny = (long)M * E, nys = (long)M * D;
    cvt_f32_bf16<<<(int)((ny / 4 + 255) / 256), 256, 0, stream>>>(y, yBf, ny);
    cvt_f32_bf16<<<(int)((nys / 4 + 255) / 256), 256, 0, stream>>>(y_states, ysBf, nys);
  }
  auto pack = [&](const float* W, long ldw, long kOff, int Kseg, int Nrows,
                  unsigned short* dst) {
    int total = (Nrows / 16) * (Kseg / 32) * 32;
    pack_w_bf16<<<(total + 255) / 256, 256, 0, stream>>>(W, ldw, kOff, Kseg, Nrows, dst);
  };
  pack(Wih0, E,    0,    E,    TH, Wih0p);
  pack(Whh0, H,    0,    H,    TH, Whh0p);
  pack(Wih1, H,    0,    H,    TH, Wih1p);
  pack(Whh1, H,    0,    H,    TH, Whh1p);
  pack(oWih, 3072, 0,    2048, TH, WcP);    // cols for [y_states, rnn_out]
  pack(oWih, 3072, 2048, H,    TH, WpP);    // cols for prev prediction
  pack(oWhh, H,    0,    H,    TH, WhhOp);
  pack(pW,   H,    0,    H,    D,  projP);

  dim3 gemmBlk(128);
  dim3 stepGrid(16), stepBlk(128);

  // ---- Phase A: GI0 = y @ Wih0^T + bih0 ----
  {
    dim3 grid(TH / 64, M / 32);
    gemm_bias_wmma<<<grid, gemmBlk, 0, stream>>>(
        yBf, E, nullptr, 0, E, E, Wih0p, bih0, GI0, TH);
  }

  // ---- Phase B: 2-layer GRU recurrence ----
  int cur = 0;
  for (int t = 0; t < T; ++t) {
    gru_step_wmma<false><<<stepGrid, stepBlk, 0, stream>>>(
        t, T, GI0,
        nullptr, 0, nullptr, nullptr,                      // layer0: gi precomputed
        h0f[cur], h0b[cur], Whh0p, bhh0,
        h0f[1 - cur], h0b[1 - cur], nullptr);
    gru_step_wmma<true><<<stepGrid, stepBlk, 0, stream>>>(
        t, T, nullptr,
        h0b[1 - cur], H, Wih1p, bih1,                      // layer1: fused input GEMM
        h1f[cur], h1b[cur], Whh1p, bhh1,
        h1f[1 - cur], h1b[1 - cur], rnnOutBf);
    cur ^= 1;
  }

  // ---- Phase C: GIc = [y_states, rnn_out] @ Wih_c^T + out_bih (dual-A) ----
  {
    dim3 grid(TH / 64, M / 32);
    gemm_bias_wmma<<<grid, gemmBlk, 0, stream>>>(
        ysBf, D, rnnOutBf, H, /*Ksplit=*/D, /*K=*/2048, WcP, obih, GIc, TH);
  }

  // ---- Phase D: output GRU with prev-prediction feedback + projection ----
  cur = 0;
  for (int t = 0; t < T; ++t) {
    const unsigned short* X = (t == 0) ? zbufBf : predBf;
    long xs = (t == 0) ? 0 : (long)D;
    gru_step_wmma<true><<<stepGrid, stepBlk, 0, stream>>>(
        t, T, GIc,
        X, xs, WpP, nullptr,                               // prev-feedback GEMM
        hof[cur], hob[cur], WhhOp, obhh,
        hof[1 - cur], hob[1 - cur], nullptr);
    proj_tanh_wmma<<<stepGrid, stepBlk, 0, stream>>>(
        t, T, hob[1 - cur], projP, pb, out, predBf);
    cur ^= 1;
  }
}
